// AttentionModule_64699387347008
// MI455X (gfx1250) — matrix operand order repaired and checked by reading, not compiled, not run on hardware
//
#include <hip/hip_runtime.h>

// ---------------- problem constants (from reference) ----------------
#define BATCH 2
#define SEQ   2048
#define DMODEL 2048
#define NHEAD 16
#define HDIM  128          // head dim
#define QW    16           // queries per wave (one WMMA M-tile)
#define WAVES 8            // waves per block
#define QBLK  (QW*WAVES)   // 128 queries per block
#define KBLK  32           // keys per iteration (two 16-wide N tiles)
#define NEGBIG (-1e9f)

typedef __attribute__((ext_vector_type(16))) __bf16 v16bf;
typedef __attribute__((ext_vector_type(8)))  __bf16 bf16x8;
typedef __attribute__((ext_vector_type(8)))  float  v8f;

// LDS pitches in bf16 elements (padded: 16B-aligned rows, bank-staggered)
#define KP 136   // K tile: 32 rows x 128 (+8 pad)
#define VP 40    // V^T tile: 128 rows x 32 (+8 pad)
#define PP 40    // P scratch per wave: 16 rows x 32 (+8 pad)

__global__ __launch_bounds__(256)
void attn_fa_bf16_wmma(const float* __restrict__ q,
                       const float* __restrict__ k,
                       const float* __restrict__ v,
                       const int*   __restrict__ kmask,
                       const int*   __restrict__ qmask,
                       float*       __restrict__ out)
{
    __shared__ __bf16 sK [KBLK*KP];        // K tile, row = key, col = d
    __shared__ __bf16 sVt[HDIM*VP];        // V tile transposed, row = d, col = key
    __shared__ __bf16 sP [WAVES*QW*PP];    // per-wave P staging

    const int tid  = threadIdx.x;
    const int lane = tid & 31;
    const int wave = tid >> 5;
    const int nL   = lane & 15;            // N index within a 16-wide tile
    const int half = lane >> 4;            // 0: lanes 0-15, 1: lanes 16-31

    const int nQB  = SEQ / QBLK;           // 16
    const int bid  = blockIdx.x;
    const int bb   = bid / (NHEAD * nQB);
    const int head = (bid / nQB) % NHEAD;
    const int q0   = (bid % nQB) * QBLK + wave * QW;

    const float scale = 0.08838834764831845f;  // 1/sqrt(128)

    // ---- Q A-fragments: 4 x (16x32) covering D=128, loaded once per wave.
    // A layout (16-bit, 16x32): lane m = lane&15; e<8 -> K=8*half+e ; e>=8 -> K=16+8*half+(e-8)
    v16bf qf[4];
    {
        const float* qrow = q + ((size_t)(bb*SEQ + q0 + nL))*DMODEL + head*HDIM;
        #pragma unroll
        for (int db = 0; db < 4; ++db) {
            union { v16bf vv; __bf16 e[16]; } u;
            #pragma unroll
            for (int e = 0; e < 8; ++e)
                u.e[e] = (__bf16)qrow[db*32 + 8*half + e];
            #pragma unroll
            for (int e = 0; e < 8; ++e)
                u.e[8+e] = (__bf16)qrow[db*32 + 16 + 8*half + e];
            qf[db] = u.vv;
        }
    }

    // query-mask bias for the 8 rows this lane's C registers hold (m = vv + 8*half)
    float qb[8];
    #pragma unroll
    for (int vv = 0; vv < 8; ++vv)
        qb[vv] = qmask[bb*SEQ + q0 + vv + 8*half] ? 0.0f : NEGBIG;

    // online softmax state + output accumulators (8 d-chunks of 16)
    float mrow[8], lrow[8];
    v8f xacc[8] = {};
    #pragma unroll
    for (int vv = 0; vv < 8; ++vv) { mrow[vv] = -3.0e38f; lrow[vv] = 0.0f; }

    __bf16* pscr = sP + wave*QW*PP;

    for (int j = 0; j < SEQ; j += KBLK) {
        // ---- cooperative stage: K tile row-major, V tile transposed (both bf16)
        {
            const int r  = tid >> 3;        // key row 0..31
            const int c0 = (tid & 7) * 16;  // 16 consecutive d
            const float* krow = k + ((size_t)(bb*SEQ + j + r))*DMODEL + head*HDIM + c0;
            const float* vrow = v + ((size_t)(bb*SEQ + j + r))*DMODEL + head*HDIM + c0;
            union { bf16x8 h[2]; __bf16 e[16]; } uk;
            __bf16 ve[16];
            #pragma unroll
            for (int e = 0; e < 16; ++e) {
                uk.e[e] = (__bf16)krow[e];
                ve[e]   = (__bf16)vrow[e];
            }
            *(bf16x8*)(&sK[r*KP + c0])     = uk.h[0];
            *(bf16x8*)(&sK[r*KP + c0 + 8]) = uk.h[1];
            #pragma unroll
            for (int e = 0; e < 16; ++e)
                sVt[(c0 + e)*VP + r] = ve[e];
        }
        __syncthreads();

        // ---- S = Q K^T for the two 16-key tiles (B layout: e<8 -> K=8*half+e ; e>=8 -> K=16+8*half+(e-8))
        v8f s0 = {}, s1 = {};
        #pragma unroll
        for (int db = 0; db < 4; ++db) {
            union { v16bf vv; bf16x8 h[2]; } b0, b1;
            b0.h[0] = *(const bf16x8*)(&sK[nL*KP      + db*32 + 8*half]);
            b0.h[1] = *(const bf16x8*)(&sK[nL*KP      + db*32 + 16 + 8*half]);
            b1.h[0] = *(const bf16x8*)(&sK[(16+nL)*KP + db*32 + 8*half]);
            b1.h[1] = *(const bf16x8*)(&sK[(16+nL)*KP + db*32 + 16 + 8*half]);
            s0 = __builtin_amdgcn_wmma_f32_16x16x32_bf16(false, qf[db], false, b0.vv,
                                                         (short)0, s0, false, false);
            s1 = __builtin_amdgcn_wmma_f32_16x16x32_bf16(false, qf[db], false, b1.vv,
                                                         (short)0, s1, false, false);
        }

        const float kb0 = kmask[bb*SEQ + j + nL]      ? 0.0f : NEGBIG;
        const float kb1 = kmask[bb*SEQ + j + 16 + nL] ? 0.0f : NEGBIG;

        // ---- masked online softmax per row (row m = vv + 8*half lives in one 16-lane half)
        #pragma unroll
        for (int vv = 0; vv < 8; ++vv) {
            float a = s0[vv]*scale + qb[vv] + kb0;
            float c = s1[vv]*scale + qb[vv] + kb1;
            float bm = fmaxf(a, c);
            #pragma unroll
            for (int mk = 1; mk <= 8; mk <<= 1)
                bm = fmaxf(bm, __shfl_xor(bm, mk, 32));
            float newm = fmaxf(mrow[vv], bm);
            float corr = __expf(mrow[vv] - newm);
            float p0   = __expf(a - newm);
            float p1   = __expf(c - newm);
            float rs = p0 + p1;
            #pragma unroll
            for (int mk = 1; mk <= 8; mk <<= 1)
                rs += __shfl_xor(rs, mk, 32);
            lrow[vv] = lrow[vv]*corr + rs;
            mrow[vv] = newm;
            #pragma unroll
            for (int dc = 0; dc < 8; ++dc) xacc[dc][vv] *= corr;
            const int m = vv + 8*half;
            pscr[m*PP + nL]      = (__bf16)p0;   // key col = nL
            pscr[m*PP + 16 + nL] = (__bf16)p1;   // key col = 16+nL
        }
        // intra-wave LDS RAW fence (cross-lane via P scratch, no block barrier needed)
        asm volatile("s_wait_dscnt 0" ::: "memory");

        // ---- P A-fragment (16x32 over the 32 keys of this block)
        union { v16bf vv; bf16x8 h[2]; } pf;
        pf.h[0] = *(const bf16x8*)(&pscr[nL*PP + 8*half]);
        pf.h[1] = *(const bf16x8*)(&pscr[nL*PP + 16 + 8*half]);

        // ---- X += P * V  (B from transposed V: row = d, col = key)
        #pragma unroll
        for (int dc = 0; dc < 8; ++dc) {
            union { v16bf vv; bf16x8 h[2]; } vf;
            vf.h[0] = *(const bf16x8*)(&sVt[(dc*16 + nL)*VP + 8*half]);
            vf.h[1] = *(const bf16x8*)(&sVt[(dc*16 + nL)*VP + 16 + 8*half]);
            xacc[dc] = __builtin_amdgcn_wmma_f32_16x16x32_bf16(false, pf.vv, false, vf.vv,
                                                               (short)0, xacc[dc], false, false);
        }
        __syncthreads();   // protect sK/sVt before next iteration's staging
    }

    // ---- finalize: divide by softmax denominator, write fp32 output
    #pragma unroll
    for (int vv = 0; vv < 8; ++vv) {
        const int m = vv + 8*half;
        const float inv = 1.0f / lrow[vv];
        float* orow = out + ((size_t)(bb*SEQ + q0 + m))*DMODEL + head*HDIM;
        #pragma unroll
        for (int dc = 0; dc < 8; ++dc)
            orow[dc*16 + nL] = xacc[dc][vv] * inv;
    }
}

extern "C" void kernel_launch(void* const* d_in, const int* in_sizes, int n_in,
                              void* d_out, int out_size, void* d_ws, size_t ws_size,
                              hipStream_t stream) {
    (void)in_sizes; (void)n_in; (void)out_size; (void)d_ws; (void)ws_size;
    const float* q     = (const float*)d_in[0];
    const float* k     = (const float*)d_in[1];
    const float* v     = (const float*)d_in[2];
    const int*   kmask = (const int*)  d_in[3];
    const int*   qmask = (const int*)  d_in[4];
    float* out = (float*)d_out;

    dim3 grid(BATCH * NHEAD * (SEQ / QBLK));   // 2*16*16 = 512 blocks
    dim3 block(256);                            // 8 wave32 waves
    attn_fa_bf16_wmma<<<grid, block, 0, stream>>>(q, k, v, kmask, qmask, out);
}
